// LSTMAttentionModel_53755810677099
// MI455X (gfx1250) — matrix operand order repaired and checked by reading, not compiled, not run on hardware
//
#include <hip/hip_runtime.h>
#include <hip/hip_bf16.h>

// ---------------------------------------------------------------------------
// LSTM + seq-attention + FC for MI455X (gfx1250, wave32, WMMA 16x16x32 f16).
// B=128, T=800, I=300, H=32, 4H=128, BT=102400.
//
// d_ws: gxp [T][8][8] frag f32 (52.4MB) | hs f32 [B,T,H] (13.1MB) |
//       hsT f16 [B][H][T] (6.6MB) | wwpack f16 (1.28MB) | wpack f16 (80KB)
// d_out: out_fc [128] f32 | qk [B,T,H] f32
// ---------------------------------------------------------------------------

typedef _Float16 v16h __attribute__((ext_vector_type(16)));
typedef _Float16 v8h  __attribute__((ext_vector_type(8)));
typedef float    v8f  __attribute__((ext_vector_type(8)));

#define BB   128
#define TT   800
#define II   300
#define HH   32
#define G4H  128
#define BT   102400

// WMMA 16x16x32 f16 fragment index maps (wave32): kA, mC per ISA 7.12.2, kB = contiguous k:
__device__ __forceinline__ int kA(int e, int hi) { return (e < 8 ? e : e + 8) + hi * 8; }
__device__ __forceinline__ int kB(int e, int hi) { return e + hi * 16; }
__device__ __forceinline__ int mC(int e, int hi) { return e + hi * 8; }

__device__ __forceinline__ float sigf(float x) {
    return 1.0f / (1.0f + __expf(-x));
}
__device__ __forceinline__ float tanh_fast(float x) {
    float a = fabsf(x);
    float e = __expf(-2.0f * a);
    float t = (1.0f - e) / (1.0f + e);
    return copysignf(t, x);
}

// ---------------------------------------------------------------------------
// Pack W_ih [128, 300] -> B-fragments f16, zero-padded K=300->320.
// wpack[((nt*10 + slab)*32 + lane)*16 + e],  nt<8, slab<10.
// ---------------------------------------------------------------------------
__global__ __launch_bounds__(256) void pack_wih_kernel(
    const float* __restrict__ W_ih, _Float16* __restrict__ wpack)
{
    const int idx  = blockIdx.x * 256 + threadIdx.x;   // < 40960
    const int e    = idx & 15;
    const int lane = (idx >> 4) & 31;
    const int l3   = idx >> 9;
    const int slab = l3 % 10;
    const int nt   = l3 / 10;
    const int hi   = (lane >> 4) & 1;
    const int n    = nt * 16 + (lane & 15);
    const int kk   = slab * 32 + kA(e, hi);
    const float v  = W_ih[n * II + (kk < II ? kk : 0)];   // unconditional load
    wpack[idx] = (kk < II) ? (_Float16)v : (_Float16)0.0f; // value select only
}

// ---------------------------------------------------------------------------
// Pack Ww [800, 800] -> A-fragments f16 (all K in-bounds).
// wwpack[((st*25 + slab)*32 + lane)*16 + e],  st<50, slab<25.
// ---------------------------------------------------------------------------
__global__ __launch_bounds__(256) void pack_ww_kernel(
    const float* __restrict__ Ww, _Float16* __restrict__ wwpack)
{
    const int idx  = blockIdx.x * 256 + threadIdx.x;   // < 640000
    const int e    = idx & 15;
    const int lane = (idx >> 4) & 31;
    const int l3   = idx >> 9;
    const int slab = l3 % 25;
    const int st   = l3 / 25;
    const int hi   = (lane >> 4) & 1;
    const int s    = st * 16 + (lane & 15);
    const int kk   = slab * 32 + kB(e, hi);
    wwpack[idx] = (_Float16)Ww[(size_t)s * TT + kk];
}

// ---------------------------------------------------------------------------
// Kernel 1: gx = x @ W_ih^T + (b_ih + b_hh), fragment-packed output.
// M is tiled as (t, batch-group): m-tile mt -> t = mt>>3, bg = mt&7,
// rows = batches bg*16..bg*16+15 at time t (matches LSTM consumer tiles).
// gxp fragment f = (t*8 + bg)*8 + nt : 32 lanes x 8 f32, lane-contiguous.
// Grid = 6400/4 blocks x 128 threads (4 waves, 1 m-tile each).
// ---------------------------------------------------------------------------
__global__ __launch_bounds__(128) void gx_gemm_kernel(
    const float* __restrict__ x, const _Float16* __restrict__ wpack,
    const float* __restrict__ b_ih, const float* __restrict__ b_hh,
    float* __restrict__ gxp)
{
    const int lane = threadIdx.x & 31;
    const int wave = threadIdx.x >> 5;
    const int ln15 = lane & 15;
    const int hi   = (lane >> 4) & 1;
    const int mt   = blockIdx.x * 4 + wave;
    const int t    = mt >> 3;
    const int bg   = mt & 7;

    v8f acc[8];
    #pragma unroll
    for (int nt = 0; nt < 8; ++nt)
        #pragma unroll
        for (int e = 0; e < 8; ++e) acc[nt][e] = 0.0f;

    // A row for this lane: x[bg*16 + ln15, t, :]
    const float* xrow = x + ((size_t)(bg * 16 + ln15) * TT + t) * II;

    // 9 full K-slabs: unconditional aligned float4 loads
    #pragma unroll 1
    for (int slab = 0; slab < 9; ++slab) {
        const int k0 = slab * 32;
        const float4 p0 = *(const float4*)(xrow + k0 + hi * 8);
        const float4 p1 = *(const float4*)(xrow + k0 + hi * 8 + 4);
        const float4 p2 = *(const float4*)(xrow + k0 + 16 + hi * 8);
        const float4 p3 = *(const float4*)(xrow + k0 + 16 + hi * 8 + 4);
        v16h a;
        a[0] = (_Float16)p0.x;  a[1] = (_Float16)p0.y;
        a[2] = (_Float16)p0.z;  a[3] = (_Float16)p0.w;
        a[4] = (_Float16)p1.x;  a[5] = (_Float16)p1.y;
        a[6] = (_Float16)p1.z;  a[7] = (_Float16)p1.w;
        a[8] = (_Float16)p2.x;  a[9] = (_Float16)p2.y;
        a[10] = (_Float16)p2.z; a[11] = (_Float16)p2.w;
        a[12] = (_Float16)p3.x; a[13] = (_Float16)p3.y;
        a[14] = (_Float16)p3.z; a[15] = (_Float16)p3.w;
        #pragma unroll
        for (int nt = 0; nt < 8; ++nt) {
            const v16h bf = *(const v16h*)(wpack + (size_t)((nt * 10 + slab) * 32 + lane) * 16);
            acc[nt] = __builtin_amdgcn_wmma_f32_16x16x32_f16(
                false, a, false, bf, (short)0, acc[nt], false, false);
        }
    }
    // tail slab (K = 288..299): branchless clamp + value select
    {
        v16h a;
        #pragma unroll
        for (int e = 0; e < 16; ++e) {
            const int kk = 288 + kA(e, hi);
            const float v = xrow[kk < II ? kk : 0];
            a[e] = (kk < II) ? (_Float16)v : (_Float16)0.0f;
        }
        #pragma unroll
        for (int nt = 0; nt < 8; ++nt) {
            const v16h bf = *(const v16h*)(wpack + (size_t)((nt * 10 + 9) * 32 + lane) * 16);
            acc[nt] = __builtin_amdgcn_wmma_f32_16x16x32_f16(
                false, a, false, bf, (short)0, acc[nt], false, false);
        }
    }

    // bias + fragment-packed store (lane-contiguous -> 2x b128 per nt)
    #pragma unroll
    for (int nt = 0; nt < 8; ++nt) {
        const float bias = b_ih[nt * 16 + ln15] + b_hh[nt * 16 + ln15];
        float* dst = gxp + ((size_t)(t * 8 + bg) * 8 + nt) * 256 + lane * 8;
        #pragma unroll
        for (int e = 0; e < 8; ++e) dst[e] = acc[nt][e] + bias;
    }
}

// ---------------------------------------------------------------------------
// Kernel 2: persistent LSTM scan. Grid = 8 blocks (16 batches), 8 waves.
// Per step/wave: C-fragment streamed from packed gx (2x b128), A from LDS
// hidden (2x ds_load_b128), 1 WMMA, gates stored fragment-ordered
// (2x ds_store_b128), then elementwise cell update.
// ---------------------------------------------------------------------------
__global__ __launch_bounds__(256) void lstm_scan_kernel(
    const float* __restrict__ gxp, const float* __restrict__ W_hh,
    float* __restrict__ hs)
{
    __shared__ _Float16 h_sh[16][HH];     // hidden state, f16 (A operand)
    __shared__ float    c_sh[16][HH];     // cell state
    __shared__ float    gfr[8][32][8];    // gate pre-acts, fragment order

    const int tid  = threadIdx.x;
    const int lane = tid & 31;
    const int wave = tid >> 5;
    const int ln15 = lane & 15;
    const int hi   = (lane >> 4) & 1;
    const int b0   = blockIdx.x * 16;
    const int n    = wave * 16 + ln15;

    #pragma unroll
    for (int j = 0; j < 2; ++j) {
        const int idx = tid + j * 256;
        ((_Float16*)h_sh)[idx] = (_Float16)0.0f;
        ((float*)c_sh)[idx]    = 0.0f;
    }

    // resident W_hh^T B-fragment: B[k,n] = W_hh[n,k]
    v16h wb;
    #pragma unroll
    for (int e = 0; e < 16; ++e) wb[e] = (_Float16)W_hh[n * HH + kA(e, hi)];

    const float* gsrc = gxp + ((size_t)blockIdx.x * 8 + wave) * 256 + lane * 8;
    __syncthreads();

    for (int t = 0; t < TT; ++t) {
        // A fragment: two contiguous 8-half runs from shared hidden
        const v8h alo = *(const v8h*)(&h_sh[ln15][hi * 8]);
        const v8h ahi = *(const v8h*)(&h_sh[ln15][16 + hi * 8]);
        v16h a;
        #pragma unroll
        for (int e = 0; e < 8; ++e) { a[e] = alo[e]; a[e + 8] = ahi[e]; }

        // C operand: packed gx fragment for (t, bg, wave)
        const float* gt = gsrc + (size_t)t * (8 * 8 * 256);
        v8f acc;
        #pragma unroll
        for (int e = 0; e < 8; ++e) acc[e] = gt[e];

        acc = __builtin_amdgcn_wmma_f32_16x16x32_f16(
            false, a, false, wb, (short)0, acc, false, false);

        #pragma unroll
        for (int e = 0; e < 8; ++e) gfr[wave][lane][e] = acc[e];
        __syncthreads();

        // cell update: 512 elems / 256 threads; fragment-order gate reads
        #pragma unroll
        for (int j = 0; j < 2; ++j) {
            const int idx = tid + j * 256;
            const int bi  = idx >> 5;       // batch in group (== matrix row m)
            const int hh  = idx & 31;       // hidden index
            const int fl  = (hh & 15) | ((bi & 8) << 1);  // fragment lane
            const int fe  = bi & 7;                        // fragment elem
            const float ig = sigf(gfr[(hh >> 4)][fl][fe]);
            const float fg = sigf(gfr[2 + (hh >> 4)][fl][fe]);
            const float gg = tanh_fast(gfr[4 + (hh >> 4)][fl][fe]);
            const float og = sigf(gfr[6 + (hh >> 4)][fl][fe]);
            const float c  = fg * c_sh[bi][hh] + ig * gg;
            const float h  = og * tanh_fast(c);
            c_sh[bi][hh] = c;
            h_sh[bi][hh] = (_Float16)h;
            hs[((size_t)(b0 + bi) * TT + t) * HH + hh] = h;  // coalesced
        }
        __syncthreads();
    }
}

// ---------------------------------------------------------------------------
// Kernel T: hs [b][t][h] f32 -> hsT [b][h][t] f16 via LDS tile transpose.
// Grid = 128 (one batch per block), 256 threads.
// ---------------------------------------------------------------------------
__global__ __launch_bounds__(256) void transpose_h_kernel(
    const float* __restrict__ hs, _Float16* __restrict__ hsT)
{
    __shared__ float tile[32][33];
    const int b   = blockIdx.x;
    const int tid = threadIdx.x;

    #pragma unroll 1
    for (int tc = 0; tc < 25; ++tc) {
        const int t0 = tc * 32;
        #pragma unroll
        for (int j = 0; j < 4; ++j) {
            const int tl = j * 8 + (tid >> 5);
            const int h  = tid & 31;
            tile[tl][h] = hs[((size_t)b * TT + t0 + tl) * HH + h];
        }
        __syncthreads();
        #pragma unroll
        for (int j = 0; j < 4; ++j) {
            const int h  = j * 8 + (tid >> 5);
            const int tl = tid & 31;
            hsT[((size_t)b * HH + h) * TT + t0 + tl] = (_Float16)tile[tl][h];
        }
        __syncthreads();
    }
}

// ---------------------------------------------------------------------------
// Kernel 3: q = hs @ Ww^T (+bw), qk = hs * q.  Per-wave (s-tile, b):
// A = packed Ww fragment (2x b128), B = hsT f16 contiguous (2x b128 each,
// two h-tiles), 25 K-slabs, 2 WMMA per slab. Grid = (50, 128) x 32.
// ---------------------------------------------------------------------------
__global__ __launch_bounds__(32) void attn_qk_kernel(
    const float* __restrict__ hs, const _Float16* __restrict__ hsT,
    const _Float16* __restrict__ wwpack, const float* __restrict__ bw,
    float* __restrict__ qk)
{
    const int lane = threadIdx.x & 31;
    const int ln15 = lane & 15;
    const int hi   = (lane >> 4) & 1;
    const int st   = blockIdx.x;
    const int s0   = st * 16;
    const int b    = blockIdx.y;

    v8f acc0, acc1;
    #pragma unroll
    for (int e = 0; e < 8; ++e) { acc0[e] = 0.0f; acc1[e] = 0.0f; }

    const _Float16* hrow0 = hsT + ((size_t)b * HH + ln15) * TT;        // h = ln15
    const _Float16* hrow1 = hsT + ((size_t)b * HH + 16 + ln15) * TT;   // h = 16+ln15
    const _Float16* wsrc  = wwpack + (size_t)(st * 25) * 512 + (size_t)lane * 16;

    #pragma unroll 1
    for (int slab = 0; slab < 25; ++slab) {
        const int t0 = slab * 32;
        const v16h a   = *(const v16h*)(wsrc + (size_t)slab * 512);
        const v16h bf0 = *(const v16h*)(hrow0 + t0 + hi * 16);
        const v16h bf1 = *(const v16h*)(hrow1 + t0 + hi * 16);
        acc0 = __builtin_amdgcn_wmma_f32_16x16x32_f16(
            false, a, false, bf0, (short)0, acc0, false, false);
        acc1 = __builtin_amdgcn_wmma_f32_16x16x32_f16(
            false, a, false, bf1, (short)0, acc1, false, false);
    }

    const float* hsb = hs + (size_t)b * TT * HH;
    #pragma unroll
    for (int e = 0; e < 8; ++e) {
        const int s = s0 + mC(e, hi);
        const float bws = bw[s];
        const float h0 = hsb[(size_t)s * HH + ln15];
        const float h1 = hsb[(size_t)s * HH + 16 + ln15];
        qk[((size_t)b * TT + s) * HH + ln15]      = h0 * (acc0[e] + bws);
        qk[((size_t)b * TT + s) * HH + 16 + ln15] = h1 * (acc1[e] + bws);
    }
}

// ---------------------------------------------------------------------------
// Kernel 4: out_fc[b] = sum_j qk_flat[b,j] * Wfc[j] + bfc (deterministic).
// ---------------------------------------------------------------------------
__global__ __launch_bounds__(256) void fc_reduce_kernel(
    const float* __restrict__ qk, const float* __restrict__ Wfc,
    const float* __restrict__ bfc, float* __restrict__ out_fc)
{
    __shared__ float red[256];
    const int b   = blockIdx.x;
    const int tid = threadIdx.x;
    const float* q = qk + (size_t)b * (TT * HH);

    float s = 0.0f;
    for (int j = tid; j < TT * HH; j += 256) s += q[j] * Wfc[j];
    red[tid] = s;
    __syncthreads();
    #pragma unroll
    for (int off = 128; off > 0; off >>= 1) {
        if (tid < off) red[tid] += red[tid + off];
        __syncthreads();
    }
    if (tid == 0) out_fc[b] = red[0] + bfc[0];
}

// ---------------------------------------------------------------------------
extern "C" void kernel_launch(void* const* d_in, const int* in_sizes, int n_in,
                              void* d_out, int out_size, void* d_ws, size_t ws_size,
                              hipStream_t stream) {
    const float* x    = (const float*)d_in[0];
    const float* W_ih = (const float*)d_in[1];
    const float* W_hh = (const float*)d_in[2];
    const float* b_ih = (const float*)d_in[3];
    const float* b_hh = (const float*)d_in[4];
    const float* Ww   = (const float*)d_in[5];
    const float* bw   = (const float*)d_in[6];
    const float* Wfc  = (const float*)d_in[7];
    const float* bfc  = (const float*)d_in[8];

    float* out    = (float*)d_out;
    float* out_fc = out;          // [128]
    float* qk     = out + BB;     // [B,T,H]

    // workspace carve-up
    float*     gxp    = (float*)d_ws;                          // 52.4 MB
    float*     hs     = gxp + (size_t)BT * G4H;                // 13.1 MB
    _Float16*  hsT    = (_Float16*)(hs + (size_t)BB * TT * HH);// 6.6 MB
    _Float16*  wwpack = hsT + (size_t)BB * HH * TT;            // 1.28 MB
    _Float16*  wpack  = wwpack + (size_t)50 * 25 * 512;        // 80 KB

    pack_wih_kernel<<<160, 256, 0, stream>>>(W_ih, wpack);
    pack_ww_kernel<<<2500, 256, 0, stream>>>(Ww, wwpack);
    gx_gemm_kernel<<<1600, 128, 0, stream>>>(x, wpack, b_ih, b_hh, gxp);
    lstm_scan_kernel<<<BB / 16, 256, 0, stream>>>(gxp, W_hh, hs);
    transpose_h_kernel<<<BB, 256, 0, stream>>>(hs, hsT);
    attn_qk_kernel<<<dim3(TT / 16, BB), 32, 0, stream>>>(hs, hsT, wwpack, bw, qk);
    fc_reduce_kernel<<<BB, 256, 0, stream>>>(qk, Wfc, bfc, out_fc);
}
